// RBFLayer_43319040148015
// MI455X (gfx1250) — matrix operand order repaired and checked by reading, not compiled, not run on hardware
//
#include <hip/hip_runtime.h>
#include <hip/hip_bf16.h>

typedef __attribute__((ext_vector_type(16))) __bf16 v16bf;
typedef __attribute__((ext_vector_type(8)))  float  v8f;

#define B_ROWS 16384
#define D_DIM  1024
#define C_COLS 2048

#define BM 128
#define BN 128
#define BK 32
#define NKT (D_DIM / BK)            // 32 k-steps
#define LDS_ROW_BYTES 80            // 32 bf16 = 64B, padded to 80B (16B aligned, spreads banks)
#define TILE_BYTES (128 * LDS_ROW_BYTES)

// ---------------------------------------------------------------------------
// Kernel 1: squared row norms of x (B rows) and centers (C rows) -> d_ws
// ---------------------------------------------------------------------------
__global__ __launch_bounds__(256)
void rbf_row_norms_kernel(const float* __restrict__ x,
                          const float* __restrict__ centers,
                          float* __restrict__ norms) {
    __shared__ float red[256];
    const int row = blockIdx.x;
    const float* src = (row < B_ROWS)
        ? (x + (size_t)row * D_DIM)
        : (centers + (size_t)(row - B_ROWS) * D_DIM);
    const int t = threadIdx.x;
    float4 v = ((const float4*)src)[t];          // 256 threads * 4 floats = 1024 = D
    float s = v.x * v.x + v.y * v.y + v.z * v.z + v.w * v.w;
    red[t] = s;
    __syncthreads();
    for (int off = 128; off > 0; off >>= 1) {
        if (t < off) red[t] += red[t + off];
        __syncthreads();
    }
    if (t == 0) norms[row] = red[0];
}

// ---------------------------------------------------------------------------
// Kernel 2: bf16 WMMA GEMM (x . centers^T) + RBF epilogue
//   block tile: 128x128x32, 256 threads = 8 waves (2x4), wave tile 64x32
// ---------------------------------------------------------------------------
__global__ __launch_bounds__(256)
void rbf_wmma_kernel(const float* __restrict__ x,
                     const float* __restrict__ centers,
                     const float* __restrict__ gamma,
                     const float* __restrict__ norms,
                     float* __restrict__ out) {
    // [buffer][A=0/B=1][tile bytes] ; 2*2*10240 = 40 KB of the 320 KB WGP LDS
    __shared__ __align__(16) unsigned char lds[2][2][TILE_BYTES];

    const int tid  = threadIdx.x;
    const int wave = tid >> 5;          // 0..7
    const int lane = tid & 31;
    const int half = lane >> 4;         // 0: lanes 0-15, 1: lanes 16-31
    const int l16  = lane & 15;
    const int wm   = wave >> 2;         // 0..1  -> 64-row slab
    const int wn   = wave & 3;          // 0..3  -> 32-col slab
    const int m0   = blockIdx.y * BM;
    const int n0   = blockIdx.x * BN;

    // ---- global->LDS copy roles: each thread moves 16 floats of A and of B
    const int crow  = tid >> 1;         // 0..127 (tile row)
    const int chalf = tid & 1;          // which 16-float half of the 32-float k-chunk
    const float* gA = x       + (size_t)(m0 + crow) * D_DIM + chalf * 16;
    const float* gB = centers + (size_t)(n0 + crow) * D_DIM + chalf * 16;
    const unsigned ldsOff = (unsigned)crow * LDS_ROW_BYTES + (unsigned)chalf * 32;

    float4 ra[4], rb[4];

    auto loadG = [&](int kt) {
        const float4* pa = (const float4*)(gA + kt * BK);
        const float4* pb = (const float4*)(gB + kt * BK);
        #pragma unroll
        for (int i = 0; i < 4; ++i) { ra[i] = pa[i]; rb[i] = pb[i]; }
    };

    auto storeT = [&](int buf, int which, const float4* r) {
        union { __bf16 h[16]; uint4 q[2]; } u;
        const float* f = (const float*)r;
        #pragma unroll
        for (int i = 0; i < 16; ++i) u.h[i] = (__bf16)f[i];
        uint4* p = (uint4*)(&lds[buf][which][ldsOff]);
        p[0] = u.q[0];
        p[1] = u.q[1];
    };

    // A fragment (16x32 bf16): lanes 0-15 hold row M=l16, K 0-7 & 16-23;
    // lanes 16-31 hold K 8-15 & 24-31 (ISA 7.12.2 16-bit A layout).
    auto loadA = [&](int buf, int tm) -> v16bf {
        unsigned row  = (unsigned)(wm * 64 + tm * 16 + l16);
        unsigned base = row * LDS_ROW_BYTES + (unsigned)half * 16;
        union { uint4 q[2]; v16bf v; } u;
        u.q[0] = *(const uint4*)(&lds[buf][0][base]);
        u.q[1] = *(const uint4*)(&lds[buf][0][base + 32]);
        return u.v;
    };

    // B fragment (32x16 bf16), tile stored N-major in LDS: lane holds column
    // N=l16; lanes 0-15 -> K 0-7 & 16-23, lanes 16-31 -> K 8-15 & 24-31, 2 K per VGPR.
    auto loadB = [&](int buf, int tn) -> v16bf {
        unsigned row  = (unsigned)(wn * 32 + tn * 16 + l16);
        unsigned base = row * LDS_ROW_BYTES + (unsigned)half * 16;
        union { uint4 q[2]; v16bf v; } u;
        u.q[0] = *(const uint4*)(&lds[buf][1][base]);
        u.q[1] = *(const uint4*)(&lds[buf][1][base + 32]);
        return u.v;
    };

    v8f acc[4][2];
    #pragma unroll
    for (int tm = 0; tm < 4; ++tm)
        #pragma unroll
        for (int tn = 0; tn < 2; ++tn)
            acc[tm][tn] = (v8f){0.f, 0.f, 0.f, 0.f, 0.f, 0.f, 0.f, 0.f};

    // prologue
    loadG(0);
    storeT(0, 0, ra);
    storeT(0, 1, rb);
    __syncthreads();

    for (int kt = 0; kt < NKT; ++kt) {
        const int cur = kt & 1;
        const int nxt = cur ^ 1;
        if (kt + 1 < NKT) loadG(kt + 1);          // prefetch next k-chunk into regs

        v16bf bfr[2];
        bfr[0] = loadB(cur, 0);
        bfr[1] = loadB(cur, 1);
        #pragma unroll
        for (int tm = 0; tm < 4; ++tm) {
            v16bf a = loadA(cur, tm);
            #pragma unroll
            for (int tn = 0; tn < 2; ++tn) {
                acc[tm][tn] = __builtin_amdgcn_wmma_f32_16x16x32_bf16(
                    false, a, false, bfr[tn], (short)0, acc[tm][tn],
                    false, false);
            }
        }

        if (kt + 1 < NKT) {
            storeT(nxt, 0, ra);
            storeT(nxt, 1, rb);
        }
        __syncthreads();
    }

    // ---- RBF epilogue: sq = x2 - 2*dot + c2 ; out = exp(-gamma*sqrt(sq))
    const float g = gamma[0];
    const float* x2 = norms;            // [B]
    const float* c2 = norms + B_ROWS;   // [C]

    #pragma unroll
    for (int tm = 0; tm < 4; ++tm) {
        const int rowBase = m0 + wm * 64 + tm * 16 + half * 8;  // VGPR r -> M=r (+8 for upper lanes)
        #pragma unroll
        for (int tn = 0; tn < 2; ++tn) {
            const int col = n0 + wn * 32 + tn * 16 + l16;
            const float cc = c2[col];
            #pragma unroll
            for (int r = 0; r < 8; ++r) {
                const int row = rowBase + r;
                float s = x2[row] - 2.0f * acc[tm][tn][r] + cc;
                s = s < 0.0f ? 0.0f : s;
                const float dist = __builtin_sqrtf(s);
                out[(size_t)row * C_COLS + col] = __expf(-g * dist);
            }
        }
    }
}

// ---------------------------------------------------------------------------
extern "C" void kernel_launch(void* const* d_in, const int* in_sizes, int n_in,
                              void* d_out, int out_size, void* d_ws, size_t ws_size,
                              hipStream_t stream) {
    const float* x       = (const float*)d_in[0];
    const float* centers = (const float*)d_in[1];
    const float* gamma   = (const float*)d_in[2];
    float* out   = (float*)d_out;
    float* norms = (float*)d_ws;        // B + C floats = 73728 bytes

    rbf_row_norms_kernel<<<B_ROWS + C_COLS, 256, 0, stream>>>(x, centers, norms);

    dim3 grid(C_COLS / BN, B_ROWS / BM);   // (16, 128)
    rbf_wmma_kernel<<<grid, 256, 0, stream>>>(x, centers, gamma, norms, out);
}
